// TransformerBlock_64330020159865
// MI455X (gfx1250) — matrix operand order repaired and checked by reading, not compiled, not run on hardware
//
#include <hip/hip_runtime.h>
#include <hip/hip_bf16.h>

// ---------------------------------------------------------------------------
// MI455X / gfx1250 transformer block (pre-LN causal MHA + FFN), bf16 WMMA.
// All matmuls via v_wmma_f32_16x16x32_bf16 (f32 accumulate), flash-attention
// style streaming softmax in f32. Compute-bound workload (~47 GFLOP vs ~100MB
// traffic) so everything rides the WMMA pipe. Attention K/V tiles are staged
// into LDS with the CDNA5 async engine (global_load_async_to_lds_b128,
// ASYNCcnt) and shared by all 8 waves of the block.
// ---------------------------------------------------------------------------

typedef __bf16 bf16;
typedef __attribute__((ext_vector_type(8)))  __bf16 v8bf;
typedef __attribute__((ext_vector_type(16))) __bf16 v16bf;
typedef __attribute__((ext_vector_type(8)))  float  v8f;

#define DDIM  256
#define SDIM  4096
#define BDIM  4
#define HDIM  4
#define DKDIM 64
#define FFDIM 128
#define MTOT  (BDIM * SDIM)   // 16384 rows

__device__ inline v8f vzero8() {
  v8f z;
#pragma unroll
  for (int i = 0; i < 8; ++i) z[i] = 0.0f;
  return z;
}

__device__ inline v8f wmma_bf16(v16bf a, v16bf b, v8f c) {
  // D = A(16x32 bf16) * B(32x16 bf16) + C(16x16 f32)
  return __builtin_amdgcn_wmma_f32_16x16x32_bf16(false, a, false, b,
                                                 (short)0, c, false, false);
}

// A-fragment (16x32, row m = lane&15): elements 0..15 hold
// K = half*16 + {0..15} (matches B-fragment order).
__device__ inline v16bf load_a_frag(const bf16* A, int lda, int l16, int hf, int k0) {
  const bf16* p = A + (size_t)l16 * lda + k0 + hf * 16;
  v8bf lo = *(const v8bf*)p;
  v8bf hi = *(const v8bf*)(p + 8);
  return __builtin_shufflevector(lo, hi, 0, 1, 2, 3, 4, 5, 6, 7,
                                         8, 9, 10, 11, 12, 13, 14, 15);
}

// Wave-level GEMM: 16 rows x 64 cols, A[M][K] bf16 row-major,
// Bt[N][K] bf16 (i.e. B transposed so each lane reads 32B contiguous).
__device__ inline void gemm_tile_16x64(const bf16* A, int lda,
                                       const bf16* Bt, int ldb,
                                       int K, v8f acc[4], int lane) {
  int l16 = lane & 15, hf = lane >> 4;
  for (int k0 = 0; k0 < K; k0 += 32) {
    __builtin_prefetch((const void*)(A + (size_t)l16 * lda + k0 + 64), 0, 3);
    v16bf aF = load_a_frag(A, lda, l16, hf, k0);
#pragma unroll
    for (int nt = 0; nt < 4; ++nt) {
      const bf16* bp = Bt + (size_t)(nt * 16 + l16) * ldb + k0 + hf * 16;
      v16bf bF = *(const v16bf*)bp;  // B-frag: K = hf*16 + {0..15}
      acc[nt] = wmma_bf16(aF, bF, acc[nt]);
    }
  }
}

// ---------------------------------------------------------------------------
// K0: convert all weights to bf16, transposed to [N][K] layout.
// ---------------------------------------------------------------------------
__global__ void k_convert_weights(const float* __restrict__ wq, const float* __restrict__ wk,
                                  const float* __restrict__ wv, const float* __restrict__ wo,
                                  const float* __restrict__ w1, const float* __restrict__ w2,
                                  bf16* __restrict__ wqT, bf16* __restrict__ wkT,
                                  bf16* __restrict__ wvT, bf16* __restrict__ woT,
                                  bf16* __restrict__ w1T, bf16* __restrict__ w2T) {
  int i = blockIdx.x * blockDim.x + threadIdx.x;   // 131072 threads
  if (i < 65536) {
    int r = i >> 8, c = i & 255;                   // source row-major [r][c]
    wqT[c * 256 + r] = (bf16)wq[i];                // wq:  [d][h*64+dk] -> [n][d]
    wkT[c * 256 + r] = (bf16)wk[i];
    wvT[c * 256 + r] = (bf16)wv[i];
    woT[c * 256 + r] = (bf16)wo[i];                // wo:  [h*64+dk][d] -> [d][hk]
  } else if (i < 98304) {
    int j = i - 65536; int d = j >> 7, f = j & 127; // w1: [d][f] -> w1T[f][d]
    w1T[f * 256 + d] = (bf16)w1[j];
  } else if (i < 131072) {
    int j = i - 98304; int f = j >> 8, d = j & 255; // w2: [f][d] -> w2T[d][f]
    w2T[d * 128 + f] = (bf16)w2[j];
  }
}

// ---------------------------------------------------------------------------
// K1: LayerNorm (one wave per 256-elem row, 8 elems/lane), bf16 out.
// ---------------------------------------------------------------------------
__global__ void k_layernorm(const float* __restrict__ x, const float* __restrict__ gw,
                            const float* __restrict__ bw, bf16* __restrict__ out) {
  int lane = threadIdx.x & 31, wvid = threadIdx.x >> 5;
  size_t row = (size_t)blockIdx.x * 8 + wvid;
  const float* xr = x + row * DDIM + lane * 8;
  float v[8];
#pragma unroll
  for (int j = 0; j < 8; ++j) v[j] = xr[j];
  float s = 0.0f;
#pragma unroll
  for (int j = 0; j < 8; ++j) s += v[j];
#pragma unroll
  for (int m = 16; m >= 1; m >>= 1) s += __shfl_xor(s, m, 32);
  float mu = s * (1.0f / 256.0f);
  float q = 0.0f;
#pragma unroll
  for (int j = 0; j < 8; ++j) { float d = v[j] - mu; v[j] = d; q += d * d; }
#pragma unroll
  for (int m = 16; m >= 1; m >>= 1) q += __shfl_xor(q, m, 32);
  float rstd = rsqrtf(q * (1.0f / 256.0f) + 1e-3f);
  v8bf ov;
#pragma unroll
  for (int j = 0; j < 8; ++j)
    ov[j] = (bf16)(v[j] * rstd * gw[lane * 8 + j] + bw[lane * 8 + j]);
  *(v8bf*)(out + row * DDIM + lane * 8) = ov;
}

// ---------------------------------------------------------------------------
// K2: fused QKV projection GEMM. Q scaled by 1/sqrt(64). V stored transposed
// [b][h][dk][s] so attention V staging reads contiguous rows.
// ---------------------------------------------------------------------------
__global__ void k_qkv(const bf16* __restrict__ hln,
                      const bf16* __restrict__ wqT, const bf16* __restrict__ wkT,
                      const bf16* __restrict__ wvT,
                      const float* __restrict__ bq, const float* __restrict__ bk,
                      const float* __restrict__ bv,
                      bf16* __restrict__ Q, bf16* __restrict__ Kc, bf16* __restrict__ Vt) {
  int wid = blockIdx.x * 8 + (threadIdx.x >> 5);   // 12288 waves
  int lane = threadIdx.x & 31;
  int which = wid / 4096;                          // 0=q 1=k 2=v
  int rem = wid % 4096;
  int m0 = (rem >> 2) * 16;
  int n0 = (rem & 3) * 64;
  const bf16* Bt = (which == 0) ? wqT : (which == 1) ? wkT : wvT;
  const float* bias = (which == 0) ? bq : (which == 1) ? bk : bv;
  v8f acc[4] = {vzero8(), vzero8(), vzero8(), vzero8()};
  gemm_tile_16x64(hln + (size_t)m0 * DDIM, DDIM, Bt + (size_t)n0 * DDIM, DDIM,
                  DDIM, acc, lane);
  int l16 = lane & 15, hf = lane >> 4;
  float scale = (which == 0) ? 0.125f : 1.0f;      // 1/sqrt(DK)
#pragma unroll
  for (int nt = 0; nt < 4; ++nt) {
    int n = n0 + nt * 16 + l16;
    int hh = n >> 6, dk = n & 63;
    float bb = bias[n];
#pragma unroll
    for (int r = 0; r < 8; ++r) {
      int m = m0 + r + hf * 8;
      int bi = m >> 12, s = m & 4095;
      float val = (acc[nt][r] + bb) * scale;
      if (which == 2) {
        Vt[((size_t)(bi * HDIM + hh) * DKDIM + dk) * SDIM + s] = (bf16)val;
      } else {
        bf16* dst = (which == 0) ? Q : Kc;
        dst[((size_t)(bi * HDIM + hh) * SDIM + s) * DKDIM + dk] = (bf16)val;
      }
    }
  }
}

// ---------------------------------------------------------------------------
// K3: causal flash attention. One block = 8 waves = 8 consecutive 16-query
// tiles of the same (b,h). K/V 32-key tiles are staged cooperatively into LDS
// with the CDNA5 async engine (ASYNCcnt) and shared by all 8 waves. Scores via
// 4 WMMAs (K=64), online softmax in f32 (16-lane xor reductions matching the
// C-layout halves), P staged through LDS into A-fragment layout, PV via 4
// WMMAs per block. Trip count is block-uniform; tail iterations are skipped
// per-wave behind a wave-uniform guard (EXEC all-ones at every WMMA).
// ---------------------------------------------------------------------------
__global__ void k_attn(const bf16* __restrict__ Q, const bf16* __restrict__ Kc,
                       const bf16* __restrict__ Vt, bf16* __restrict__ ctx) {
  __shared__ __attribute__((aligned(32))) bf16 lds_k[32][64];      // 4KB: [key][dk]
  __shared__ __attribute__((aligned(32))) bf16 lds_v[64][32];      // 4KB: [dk][key]
  __shared__ __attribute__((aligned(32))) bf16 lds_p[8][16][32];   // 8KB
  int tid = threadIdx.x;
  int wslot = tid >> 5;
  int lane = tid & 31;
  int blk = blockIdx.x;          // 512 blocks: 16 (b,h) x 32 block-rows
  int bh = blk >> 5;             // 0..15
  int h = bh & 3, b = bh >> 2;
  int qt = (blk & 31) * 8 + wslot;
  int q0 = qt * 16;
  int q0max = ((blk & 31) * 8 + 7) * 16;   // last wave's q0 (block-uniform)
  int l16 = lane & 15, hf = lane >> 4;
  const bf16* Qb = Q  + (size_t)bh * SDIM * DKDIM;
  const bf16* Kb = Kc + (size_t)bh * SDIM * DKDIM;
  const bf16* Vb = Vt + (size_t)bh * DKDIM * SDIM;

  v16bf aQ0 = load_a_frag(Qb + (size_t)q0 * DKDIM, DKDIM, l16, hf, 0);
  v16bf aQ1 = load_a_frag(Qb + (size_t)q0 * DKDIM, DKDIM, l16, hf, 32);

  v8f o0 = vzero8(), o1 = vzero8(), o2 = vzero8(), o3 = vzero8();
  float mrun[8], lrun[8];
#pragma unroll
  for (int r = 0; r < 8; ++r) { mrun[r] = -1e30f; lrun[r] = 1.0f; }

  // Per-thread staging slots: K tile = 32 rows x 128B (8 chunks of 16B),
  // V tile = 64 rows x 64B (4 chunks of 16B). 256 threads cover each exactly.
  unsigned koff = (unsigned)(size_t)&lds_k[tid >> 3][(tid & 7) * 8];
  unsigned voff = (unsigned)(size_t)&lds_v[tid >> 2][(tid & 3) * 8];

  for (int j0 = 0; j0 < q0max + 16; j0 += 32) {
    // ---- cooperative async staging of K/V 32-key tiles into LDS ----
    {
      const bf16* kgp = Kb + (size_t)(j0 + (tid >> 3)) * DKDIM + (tid & 7) * 8;
      const bf16* vgp = Vb + (size_t)(tid >> 2) * SDIM + j0 + (tid & 3) * 8;
      asm volatile(
          "global_load_async_to_lds_b128 %0, %1, off\n\t"
          "global_load_async_to_lds_b128 %2, %3, off\n\t"
          "s_wait_asynccnt 0x0"
          :
          : "v"(koff), "v"((unsigned long long)(size_t)kgp),
            "v"(voff), "v"((unsigned long long)(size_t)vgp)
          : "memory");
    }
    __syncthreads();

    if (j0 < q0 + 16) {   // wave-uniform: this wave's causal range
      // ---- scores: two 16x16 tiles over keys [j0, j0+32), from LDS ----
      v8f s0 = vzero8(), s1 = vzero8();
      {
        const bf16* kp0 = &lds_k[l16][hf * 16];
        s0 = wmma_bf16(aQ0, *(const v16bf*)kp0, s0);
        s0 = wmma_bf16(aQ1, *(const v16bf*)(kp0 + 32), s0);
        const bf16* kp1 = &lds_k[16 + l16][hf * 16];
        s1 = wmma_bf16(aQ0, *(const v16bf*)kp1, s1);
        s1 = wmma_bf16(aQ1, *(const v16bf*)(kp1 + 32), s1);
      }
      // ---- causal mask (diagonal blocks only; wave-uniform guard) ----
      if (j0 + 31 > q0) {
#pragma unroll
        for (int r = 0; r < 8; ++r) {
          int m = q0 + r + hf * 8;       // global query row of C element r
          int kg = j0 + l16;             // global key col (tile 0)
          if (kg > m)      s0[r] = -1e30f;
          if (kg + 16 > m) s1[r] = -1e30f;
        }
      }
      // ---- online softmax update ----
      float al[8];
#pragma unroll
      for (int r = 0; r < 8; ++r) {
        float tm = fmaxf(s0[r], s1[r]);
        tm = fmaxf(tm, __shfl_xor(tm, 1, 32));
        tm = fmaxf(tm, __shfl_xor(tm, 2, 32));
        tm = fmaxf(tm, __shfl_xor(tm, 4, 32));
        tm = fmaxf(tm, __shfl_xor(tm, 8, 32));
        float mn = fmaxf(mrun[r], tm);
        al[r] = __expf(mrun[r] - mn);
        mrun[r] = mn;
        float p0 = __expf(s0[r] - mn);
        float p1 = __expf(s1[r] - mn);
        float rs = p0 + p1;
        rs += __shfl_xor(rs, 1, 32);
        rs += __shfl_xor(rs, 2, 32);
        rs += __shfl_xor(rs, 4, 32);
        rs += __shfl_xor(rs, 8, 32);
        lrun[r] = (j0 == 0) ? rs : lrun[r] * al[r] + rs;
        int row = r + hf * 8;
        lds_p[wslot][row][l16]      = (bf16)p0;   // same-wave LDS round trip
        lds_p[wslot][row][16 + l16] = (bf16)p1;   // (C layout -> A layout)
      }
#pragma unroll
      for (int r = 0; r < 8; ++r) {
        o0[r] *= al[r]; o1[r] *= al[r]; o2[r] *= al[r]; o3[r] *= al[r];
      }
      // ---- P as A-fragment from LDS ----
      const bf16* pp = &lds_p[wslot][l16][0];
      v8bf plo = *(const v8bf*)(pp + hf * 16);
      v8bf phi = *(const v8bf*)(pp + hf * 16 + 8);
      v16bf aP = __builtin_shufflevector(plo, phi, 0, 1, 2, 3, 4, 5, 6, 7,
                                                   8, 9, 10, 11, 12, 13, 14, 15);
      // ---- PV from LDS (V stored [dk][key], 16 contiguous keys per lane) ----
      const bf16* vp = &lds_v[l16][hf * 16];
      o0 = wmma_bf16(aP, *(const v16bf*)(vp),           o0);
      o1 = wmma_bf16(aP, *(const v16bf*)(vp + 16 * 32), o1);
      o2 = wmma_bf16(aP, *(const v16bf*)(vp + 32 * 32), o2);
      o3 = wmma_bf16(aP, *(const v16bf*)(vp + 48 * 32), o3);
    }
    __syncthreads();
  }
  // ---- normalize and write ctx [b*S+s][h*64+dk] bf16 ----
  size_t base = ((size_t)b * SDIM + q0) * DDIM + h * DKDIM;
#pragma unroll
  for (int r = 0; r < 8; ++r) {
    float inv = 1.0f / lrun[r];
    int row = r + hf * 8;
    bf16* cp = ctx + base + (size_t)row * DDIM + l16;
    cp[0]  = (bf16)(o0[r] * inv);
    cp[16] = (bf16)(o1[r] * inv);
    cp[32] = (bf16)(o2[r] * inv);
    cp[48] = (bf16)(o3[r] * inv);
  }
}

// ---------------------------------------------------------------------------
// K4: output projection + residual: out = x + ctx @ Wo + bo   (f32)
// ---------------------------------------------------------------------------
__global__ void k_oproj(const bf16* __restrict__ ctx, const bf16* __restrict__ woT,
                        const float* __restrict__ bo, const float* __restrict__ x,
                        float* __restrict__ out) {
  int wid = blockIdx.x * 8 + (threadIdx.x >> 5);   // 4096 waves
  int lane = threadIdx.x & 31;
  int m0 = (wid >> 2) * 16, n0 = (wid & 3) * 64;
  v8f acc[4] = {vzero8(), vzero8(), vzero8(), vzero8()};
  gemm_tile_16x64(ctx + (size_t)m0 * DDIM, DDIM, woT + (size_t)n0 * DDIM, DDIM,
                  DDIM, acc, lane);
  int l16 = lane & 15, hf = lane >> 4;
#pragma unroll
  for (int nt = 0; nt < 4; ++nt) {
    int n = n0 + nt * 16 + l16;
    float bb = bo[n];
#pragma unroll
    for (int r = 0; r < 8; ++r) {
      size_t m = m0 + r + hf * 8;
      out[m * DDIM + n] = x[m * DDIM + n] + acc[nt][r] + bb;
    }
  }
}

// ---------------------------------------------------------------------------
// K5: FFN layer 1: a1 = relu(h2 @ W1 + b1)  -> bf16 [M][128]
// ---------------------------------------------------------------------------
__global__ void k_ffn1(const bf16* __restrict__ h2, const bf16* __restrict__ w1T,
                       const float* __restrict__ b1, bf16* __restrict__ a1) {
  int wid = blockIdx.x * 8 + (threadIdx.x >> 5);   // 2048 waves
  int lane = threadIdx.x & 31;
  int m0 = (wid >> 1) * 16, n0 = (wid & 1) * 64;
  v8f acc[4] = {vzero8(), vzero8(), vzero8(), vzero8()};
  gemm_tile_16x64(h2 + (size_t)m0 * DDIM, DDIM, w1T + (size_t)n0 * DDIM, DDIM,
                  DDIM, acc, lane);
  int l16 = lane & 15, hf = lane >> 4;
#pragma unroll
  for (int nt = 0; nt < 4; ++nt) {
    int n = n0 + nt * 16 + l16;
    float bb = b1[n];
#pragma unroll
    for (int r = 0; r < 8; ++r) {
      size_t m = m0 + r + hf * 8;
      a1[m * FFDIM + n] = (bf16)fmaxf(acc[nt][r] + bb, 0.0f);
    }
  }
}

// ---------------------------------------------------------------------------
// K6: FFN layer 2 + residual accumulate: out += a1 @ W2 + b2
// ---------------------------------------------------------------------------
__global__ void k_ffn2(const bf16* __restrict__ a1, const bf16* __restrict__ w2T,
                       const float* __restrict__ b2, float* __restrict__ out) {
  int wid = blockIdx.x * 8 + (threadIdx.x >> 5);   // 4096 waves
  int lane = threadIdx.x & 31;
  int m0 = (wid >> 2) * 16, n0 = (wid & 3) * 64;
  v8f acc[4] = {vzero8(), vzero8(), vzero8(), vzero8()};
  gemm_tile_16x64(a1 + (size_t)m0 * FFDIM, FFDIM, w2T + (size_t)n0 * FFDIM, FFDIM,
                  FFDIM, acc, lane);
  int l16 = lane & 15, hf = lane >> 4;
#pragma unroll
  for (int nt = 0; nt < 4; ++nt) {
    int n = n0 + nt * 16 + l16;
    float bb = b2[n];
#pragma unroll
    for (int r = 0; r < 8; ++r) {
      size_t m = m0 + r + hf * 8;
      out[m * DDIM + n] += acc[nt][r] + bb;
    }
  }
}

// ---------------------------------------------------------------------------
extern "C" void kernel_launch(void* const* d_in, const int* in_sizes, int n_in,
                              void* d_out, int out_size, void* d_ws, size_t ws_size,
                              hipStream_t stream) {
  const float* x     = (const float*)d_in[0];
  const float* ln1_g = (const float*)d_in[1];
  const float* ln1_b = (const float*)d_in[2];
  const float* ln2_g = (const float*)d_in[3];
  const float* ln2_b = (const float*)d_in[4];
  const float* wq    = (const float*)d_in[5];
  const float* bq    = (const float*)d_in[6];
  const float* wk    = (const float*)d_in[7];
  const float* bk    = (const float*)d_in[8];
  const float* wv    = (const float*)d_in[9];
  const float* bv    = (const float*)d_in[10];
  const float* wo    = (const float*)d_in[11];
  const float* bo    = (const float*)d_in[12];
  const float* w1    = (const float*)d_in[13];
  const float* b1    = (const float*)d_in[14];
  const float* w2    = (const float*)d_in[15];
  const float* b2    = (const float*)d_in[16];
  float* out = (float*)d_out;

  char* ws = (char*)d_ws;
  size_t off = 0;
  auto alloc = [&](size_t bytes) -> void* {
    void* p = (void*)(ws + off);
    off += (bytes + 255) & ~(size_t)255;
    return p;
  };
  bf16* hln1 = (bf16*)alloc((size_t)MTOT * DDIM * 2);
  bf16* h2   = (bf16*)alloc((size_t)MTOT * DDIM * 2);
  bf16* Qb   = (bf16*)alloc((size_t)BDIM * HDIM * SDIM * DKDIM * 2);
  bf16* Kb   = (bf16*)alloc((size_t)BDIM * HDIM * SDIM * DKDIM * 2);
  bf16* Vt   = (bf16*)alloc((size_t)BDIM * HDIM * DKDIM * SDIM * 2);
  bf16* ctx  = (bf16*)alloc((size_t)MTOT * DDIM * 2);
  bf16* a1   = (bf16*)alloc((size_t)MTOT * FFDIM * 2);
  bf16* wqT  = (bf16*)alloc((size_t)DDIM * DDIM * 2);
  bf16* wkT  = (bf16*)alloc((size_t)DDIM * DDIM * 2);
  bf16* wvT  = (bf16*)alloc((size_t)DDIM * DDIM * 2);
  bf16* woT  = (bf16*)alloc((size_t)DDIM * DDIM * 2);
  bf16* w1T  = (bf16*)alloc((size_t)FFDIM * DDIM * 2);
  bf16* w2T  = (bf16*)alloc((size_t)DDIM * FFDIM * 2);

  k_convert_weights<<<512, 256, 0, stream>>>(wq, wk, wv, wo, w1, w2,
                                             wqT, wkT, wvT, woT, w1T, w2T);
  k_layernorm<<<MTOT / 8, 256, 0, stream>>>(x, ln1_g, ln1_b, hln1);
  k_qkv<<<1536, 256, 0, stream>>>(hln1, wqT, wkT, wvT, bq, bk, bv, Qb, Kb, Vt);
  k_attn<<<512, 256, 0, stream>>>(Qb, Kb, Vt, ctx);
  k_oproj<<<512, 256, 0, stream>>>(ctx, woT, bo, x, out);
  k_layernorm<<<MTOT / 8, 256, 0, stream>>>(out, ln2_g, ln2_b, h2);
  k_ffn1<<<256, 256, 0, stream>>>(h2, w1T, b1, a1);
  k_ffn2<<<512, 256, 0, stream>>>(a1, w2T, b2, out);
}
